// Precomputation_72799695667509
// MI455X (gfx1250) — compile-verified, matrix operand order checked by reading, not run on hardware
//
#include <hip/hip_runtime.h>

typedef __attribute__((ext_vector_type(16))) _Float16 v16h;
typedef __attribute__((ext_vector_type(8)))  _Float16 v8h;
typedef __attribute__((ext_vector_type(4)))  _Float16 v4h;
typedef __attribute__((ext_vector_type(8)))  float    v8f;

// ---------------------------------------------------------------------------
// Problem constants
// ---------------------------------------------------------------------------
#define N_SNAP     128
#define N_ATOMS    256
#define AEV_DIM    384
#define N_MODELS   8
#define N_SPECIES  4
#define OUT_DIM    96
#define MAX_DIM    160

// Per-(model,species) packed-weight block layout (halves / bytes)
//   W1pack : 120 frags * 512 halves = 61440 halves   @ half ofs 0
//   W2pack :  40 frags * 512 halves = 20480 halves   @ half ofs 61440
//   W3pack :  24 frags * 512 halves = 12288 halves   @ half ofs 81920
//   b1 : 160 f32 @ byte 188416 ; b2 : 128 f32 @ 189056 ; b3 : 96 f32 @ 189568
#define WBLOCK_BYTES 189952
#define WS_COUNTS_OFS   0
#define WS_ATOMLIST_OFS 16
#define WS_WEIGHTS_OFS  4352
#define WS_AEVH_OFS     (WS_WEIGHTS_OFS + 32 * WBLOCK_BYTES)   // 6082816
// total ws use: 6082816 + 128*256*384*2 = ~31.2 MB

__device__ __forceinline__ float celu01(float x) {
    return x > 0.f ? x : 0.1f * (__expf(x * 10.f) - 1.f);
}

// A-fragment: 16-bit A 16x32 layout. Lane L holds row M=L%16; K-half = L/16.
// K order per lane: [kh*8 .. kh*8+7], then [16+kh*8 .. 16+kh*8+7].
__device__ __forceinline__ v16h load_a_frag(const _Float16* base) {
    v8h lo = *(const v8h*)(base);
    v8h hi = *(const v8h*)(base + 16);
    return __builtin_shufflevector(lo, hi, 0,1,2,3,4,5,6,7,8,9,10,11,12,13,14,15);
}

// B-fragment: pre-packed so lane L's 16 halves are contiguous (2x b128).
__device__ __forceinline__ v16h load_b_frag(const _Float16* p) {
    v8h lo = *(const v8h*)(p);
    v8h hi = *(const v8h*)(p + 8);
    return __builtin_shufflevector(lo, hi, 0,1,2,3,4,5,6,7,8,9,10,11,12,13,14,15);
}

// ---------------------------------------------------------------------------
// Kernel 1: per-species atom compaction (species row constant across snaps)
// ---------------------------------------------------------------------------
__global__ void compact_species_kernel(const int* __restrict__ species,
                                       int* __restrict__ counts,
                                       int* __restrict__ atomlist) {
    __shared__ int cnt[N_SPECIES];
    const int t = threadIdx.x;                 // 0..255 == atom index
    if (t < N_SPECIES) cnt[t] = 0;
    __syncthreads();
    const int s = species[t];                  // species[0][t]
    const int pos = atomicAdd(&cnt[s], 1);
    atomlist[s * N_ATOMS + pos] = t;
    __syncthreads();
    if (t < N_SPECIES) counts[t] = cnt[t];
}

// ---------------------------------------------------------------------------
// Kernel 2: species passthrough output (first returned tensor, cast to f32)
// ---------------------------------------------------------------------------
__global__ void species_out_kernel(const int* __restrict__ sp, float* __restrict__ out) {
    const int i = blockIdx.x * 256 + threadIdx.x;
    if (i < N_SNAP * N_ATOMS) out[i] = (float)sp[i];
}

// ---------------------------------------------------------------------------
// Kernel 3: one-time AEV f32 -> f16 conversion (read 8x afterwards as f16)
// ---------------------------------------------------------------------------
__global__ void aev_to_f16_kernel(const float* __restrict__ aev, _Float16* __restrict__ aevh) {
    const int i = blockIdx.x * 256 + threadIdx.x;   // one float4 per thread
    const float4 v = ((const float4*)aev)[i];
    v4h h; h.x = (_Float16)v.x; h.y = (_Float16)v.y;
    h.z = (_Float16)v.z; h.w = (_Float16)v.w;
    *(v4h*)(aevh + (size_t)i * 4) = h;
}

// ---------------------------------------------------------------------------
// Kernel 4: pack one (model,species) weight set into f16 B-fragment layout.
// 184 fragments * 32 lanes + 384 bias entries = 6272 work items.
// ---------------------------------------------------------------------------
__global__ void pack_weights_kernel(const float* __restrict__ W1, const float* __restrict__ b1,
                                    const float* __restrict__ W2, const float* __restrict__ b2,
                                    const float* __restrict__ W3, const float* __restrict__ b3,
                                    int h1, int h2, char* __restrict__ dstbase) {
    const int tid = blockIdx.x * 256 + threadIdx.x;
    _Float16* Wp = (_Float16*)dstbase;
    if (tid < 184 * 32) {
        const int f = tid >> 5, lane = tid & 31;
        const float* W; int in_dim, out_dim, nt, kt; _Float16* dst;
        if (f < 120) {            // layer1: 10 N-tiles x 12 K-steps
            nt = f / 12; kt = f % 12;
            W = W1; in_dim = AEV_DIM; out_dim = h1;
            dst = Wp + f * 512;
        } else if (f < 160) {     // layer2: 8 N-tiles x 5 K-steps (K padded to 160)
            const int fi = f - 120; nt = fi / 5; kt = fi % 5;
            W = W2; in_dim = h1; out_dim = h2;
            dst = Wp + 61440 + fi * 512;
        } else {                  // layer3: 6 N-tiles x 4 K-steps (K padded to 128)
            const int fi = f - 160; nt = fi / 4; kt = fi % 4;
            W = W3; in_dim = h2; out_dim = OUT_DIM;
            dst = Wp + 81920 + fi * 512;
        }
        const int n  = nt * 16 + (lane & 15);
        const int kh = lane >> 4;
        _Float16* d = dst + lane * 16;
        #pragma unroll
        for (int j = 0; j < 16; ++j) {
            const int k = kt * 32 + ((j < 8) ? (kh * 8 + j) : (16 + kh * 8 + (j - 8)));
            const float v = (n < out_dim && k < in_dim) ? W[n * in_dim + k] : 0.f;
            d[j] = (_Float16)v;
        }
    } else {
        const int i = tid - 184 * 32;
        if (i < 160)       ((float*)(dstbase + 188416))[i]       = (i < h1)        ? b1[i]       : 0.f;
        else if (i < 288)  ((float*)(dstbase + 189056))[i - 160] = (i - 160 < h2)  ? b2[i - 160] : 0.f;
        else if (i < 384)  ((float*)(dstbase + 189568))[i - 288] = b3[i - 288];
    }
}

// ---------------------------------------------------------------------------
// Kernel 5: fused 3-layer MLP. Block = 32 rows (one atom x 32 snapshots),
// 128 threads = 4 waves; wave w -> row-half (w&1)*16, N-tile parity w>>1.
// LDS: [X f16 32x384 | aliased: Out f32 32x160] + H1 f16 32x160 + H2 f16 32x128
// ---------------------------------------------------------------------------
__global__ __launch_bounds__(128)
void ani_mlp_kernel(const _Float16* __restrict__ aevh, const char* __restrict__ wsW,
                    const int* __restrict__ counts, const int* __restrict__ atomlist,
                    float* __restrict__ outE) {
    static const int H1D[N_SPECIES] = {160, 144, 128, 128};
    static const int H2D[N_SPECIES] = {128, 112, 112, 112};

    const int s = blockIdx.y, m = blockIdx.z;
    const int h1 = H1D[s], h2 = H2D[s];
    const int h1pad = (h1 + 31) & ~31;                // 160/160/128/128
    const int rowbase = blockIdx.x * 32;
    if (rowbase >= counts[s] * N_SNAP) return;
    const int atom  = atomlist[s * N_ATOMS + (rowbase >> 7)];
    const int snap0 = rowbase & (N_SNAP - 1);

    __shared__ __align__(16) char smem[43008];
    _Float16* Xh   = (_Float16*)smem;                 // 32*384 f16 = 24576 B
    float*    Outs = (float*)smem;                    // 32*160 f32 = 20480 B (aliases Xh)
    _Float16* H1s  = (_Float16*)(smem + 24576);       // 32*160 f16 = 10240 B
    _Float16* H2s  = (_Float16*)(smem + 34816);       // 32*128 f16 =  8192 B

    const int tid = threadIdx.x;

    const char* base = wsW + (size_t)(m * N_SPECIES + s) * WBLOCK_BYTES;
    __builtin_prefetch(base, 0, 1);                   // global_prefetch_b8 (warm L2)
    const _Float16* W1p = (const _Float16*)base;
    const _Float16* W2p = W1p + 61440;
    const _Float16* W3p = W1p + 81920;
    const float* b1 = (const float*)(base + 188416);
    const float* b2 = (const float*)(base + 189056);
    const float* b3 = (const float*)(base + 189568);

    // ---- stage X tile: pure f16 b128 copies (48 chunks of 8 halves per row) ----
    for (int idx = tid; idx < 32 * 48; idx += 128) {
        const int i = idx / 48, c = idx % 48;
        const v8h v = *(const v8h*)(aevh + ((size_t)(snap0 + i) * N_ATOMS + atom) * AEV_DIM + c * 8);
        *(v8h*)(&Xh[i * AEV_DIM + c * 8]) = v;
    }

    const int wave = tid >> 5, lane = tid & 31;
    const int rowHalf = (wave & 1) * 16;
    const int ntOff   = wave >> 1;
    const int mrow = lane & 15, kh = lane >> 4;
    const int arow = rowHalf + mrow;

    // zero H1 K-pad columns (s==1: cols 144..159) before compute
    {
        const int padc = h1pad - h1;
        for (int idx = tid; idx < 32 * padc; idx += 128) {
            const int i = idx / padc, cc = h1 + idx % padc;
            H1s[i * 160 + cc] = (_Float16)0.f;
        }
    }
    __syncthreads();

    // ---- layer 1: [32x384] x [384xh1] ----
    for (int nt = ntOff; nt < (h1 >> 4); nt += 2) {
        v8f acc = {};
        #pragma unroll
        for (int kt = 0; kt < 12; ++kt) {
            const v16h A = load_a_frag(&Xh[arow * AEV_DIM + kt * 32 + kh * 8]);
            const v16h B = load_b_frag(W1p + (nt * 12 + kt) * 512 + lane * 16);
            acc = __builtin_amdgcn_wmma_f32_16x16x32_f16(false, A, false, B,
                                                         (short)0, acc, false, false);
        }
        const int n = nt * 16 + mrow;
        const float bv = b1[n];
        #pragma unroll
        for (int r = 0; r < 8; ++r)
            H1s[(rowHalf + kh * 8 + r) * 160 + n] = (_Float16)celu01(acc[r] + bv);
    }
    // zero H2 K-pad columns (s>=1: cols 112..127)
    {
        const int padc = 128 - h2;
        for (int idx = tid; idx < 32 * padc; idx += 128) {
            const int i = idx / padc, cc = h2 + idx % padc;
            H2s[i * 128 + cc] = (_Float16)0.f;
        }
    }
    __syncthreads();   // X tile dead beyond this point -> Outs may reuse it later

    // ---- layer 2: [32 x h1pad] x [h1pad x h2] ----
    const int k2 = h1pad >> 5;                         // 5 or 4
    for (int nt = ntOff; nt < (h2 >> 4); nt += 2) {
        v8f acc = {};
        for (int kt = 0; kt < k2; ++kt) {
            const v16h A = load_a_frag(&H1s[arow * 160 + kt * 32 + kh * 8]);
            const v16h B = load_b_frag(W2p + (nt * 5 + kt) * 512 + lane * 16);
            acc = __builtin_amdgcn_wmma_f32_16x16x32_f16(false, A, false, B,
                                                         (short)0, acc, false, false);
        }
        const int n = nt * 16 + mrow;
        const float bv = b2[n];
        #pragma unroll
        for (int r = 0; r < 8; ++r)
            H2s[(rowHalf + kh * 8 + r) * 128 + n] = (_Float16)celu01(acc[r] + bv);
    }
    __syncthreads();

    // ---- layer 3: [32x128] x [128x96] -> stage f32 rows in LDS ----
    for (int nt = ntOff; nt < 6; nt += 2) {
        v8f acc = {};
        #pragma unroll
        for (int kt = 0; kt < 4; ++kt) {
            const v16h A = load_a_frag(&H2s[arow * 128 + kt * 32 + kh * 8]);
            const v16h B = load_b_frag(W3p + (nt * 4 + kt) * 512 + lane * 16);
            acc = __builtin_amdgcn_wmma_f32_16x16x32_f16(false, A, false, B,
                                                         (short)0, acc, false, false);
        }
        const int n = nt * 16 + mrow;
        const float bv = b3[n];
        #pragma unroll
        for (int r = 0; r < 8; ++r)
            Outs[(rowHalf + kh * 8 + r) * MAX_DIM + n] = celu01(acc[r] + bv);
    }
    // NaN tail (dims 96..159) into the staged rows
    const float nanv = __builtin_nanf("");
    for (int idx = tid; idx < 32 * 64; idx += 128) {
        const int i = idx >> 6, f = 96 + (idx & 63);
        Outs[i * MAX_DIM + f] = nanv;
    }
    __syncthreads();

    // ---- coalesced output: 32 rows x 40 float4 (640 B per row, contiguous) ----
    float4* out4 = (float4*)outE;
    for (int idx = tid; idx < 32 * 40; idx += 128) {
        const int i = idx / 40, c = idx % 40;
        const size_t rowbaseG = (((size_t)(snap0 + i) * N_ATOMS + atom) * N_MODELS + m) * MAX_DIM;
        out4[rowbaseG / 4 + c] = *(const float4*)(&Outs[i * MAX_DIM + c * 4]);
    }
}

// ---------------------------------------------------------------------------
// Host launcher
// ---------------------------------------------------------------------------
extern "C" void kernel_launch(void* const* d_in, const int* in_sizes, int n_in,
                              void* d_out, int out_size, void* d_ws, size_t ws_size,
                              hipStream_t stream) {
    (void)in_sizes; (void)n_in; (void)out_size; (void)ws_size;

    const int*   d_species = (const int*)d_in[0];
    const float* d_aev     = (const float*)d_in[1];

    char* ws        = (char*)d_ws;
    int*  counts    = (int*)(ws + WS_COUNTS_OFS);
    int*  atomlist  = (int*)(ws + WS_ATOMLIST_OFS);
    char* wbase     = ws + WS_WEIGHTS_OFS;
    _Float16* aevh  = (_Float16*)(ws + WS_AEVH_OFS);

    float* outF = (float*)d_out;                 // species (as f32), 32768 elems
    float* outE = outF + N_SNAP * N_ATOMS;       // energies [128,256,8,160]

    compact_species_kernel<<<1, 256, 0, stream>>>(d_species, counts, atomlist);
    species_out_kernel<<<(N_SNAP * N_ATOMS + 255) / 256, 256, 0, stream>>>(d_species, outF);

    // 128*256*384 / 4 floats per thread = 3,145,728 threads
    aev_to_f16_kernel<<<(N_SNAP * N_ATOMS * AEV_DIM / 4 + 255) / 256, 256, 0, stream>>>(d_aev, aevh);

    static const int H1D[N_SPECIES] = {160, 144, 128, 128};
    static const int H2D[N_SPECIES] = {128, 112, 112, 112};
    for (int m = 0; m < N_MODELS; ++m) {
        for (int s = 0; s < N_SPECIES; ++s) {
            const int bi = 2 + (m * N_SPECIES + s) * 6;
            pack_weights_kernel<<<25, 256, 0, stream>>>(
                (const float*)d_in[bi + 0], (const float*)d_in[bi + 1],
                (const float*)d_in[bi + 2], (const float*)d_in[bi + 3],
                (const float*)d_in[bi + 4], (const float*)d_in[bi + 5],
                H1D[s], H2D[s],
                wbase + (size_t)(m * N_SPECIES + s) * WBLOCK_BYTES);
        }
    }

    // grid.x covers worst case (all 256 atoms one species): 32768 rows / 32
    ani_mlp_kernel<<<dim3(1024, N_SPECIES, N_MODELS), 128, 0, stream>>>(
        aevh, wbase, counts, atomlist, outE);
}
